// Encoder_54889682043409
// MI455X (gfx1250) — matrix operand order repaired and checked by reading, not compiled, not run on hardware
//
#include <hip/hip_runtime.h>

// ---------------------------------------------------------------------------
// Transformer encoder on gfx1250 (MI455X): all GEMMs via v_wmma_f32_16x16x32_f16.
// Weights + GEMM-feeding activations kept in f16; guard-free vectorized staging,
// double-buffered LDS, software-pipelined global loads.
// ---------------------------------------------------------------------------

typedef __attribute__((ext_vector_type(16))) _Float16 v16h;
typedef __attribute__((ext_vector_type(8)))  _Float16 v8h;
typedef __attribute__((ext_vector_type(4)))  _Float16 v4h;
typedef __attribute__((ext_vector_type(8)))  float    v8f;
typedef __attribute__((ext_vector_type(4)))  float    v4f;

#define PA 40   // u16 pitch for LDS tile rows: 80B rows -> 16B-aligned v8h slots

union Frag { v16h h; unsigned int u[8]; };

// Batched WMMA GEMM, C = alpha*(A@B) + bias, optional ReLU.
//  A: [.,K] (f16 if AF16 else f32), lda;  B: f16, element (k,n) = BT ? B[n*ldb+k] : B[k*ldb+n]
//  C: f16 if OUT16 else f32. Requires: M % TBM == 0, N % TBN == 0, K % 32 == 0.
//  Batch z: zb = z/Hdiv, zh = z%Hdiv with per-level element offsets.
template <int TBM, int TBN, int WROWS, bool BT, bool RELU, bool AF16, bool OUT16>
__global__ __launch_bounds__(256) void gemm2(
    const void* __restrict__ Aptr, const _Float16* __restrict__ Bh,
    const float* __restrict__ bias, void* __restrict__ Cptr,
    int K, int lda, int ldb, int ldc,
    long long sAb, long long sAh, long long sBb, long long sBh,
    long long sCb, long long sCh, int Hdiv, float alpha)
{
    constexpr int WCOLS = 8 / WROWS;
    constexpr int MT    = TBM / (WROWS * 16);   // m-tiles per wave
    constexpr int NT    = TBN / (WCOLS * 16);   // n-tiles per wave
    constexpr int CA16  = TBM / 64;             // v8h chunks/thread for f16 A
    constexpr int CA32  = TBM / 32;             // v4f chunks/thread for f32 A
    constexpr int CB    = TBN / 64;             // v8h chunks/thread for B
    constexpr int NB8   = TBN / 8;              // 8-wide chunks per B row

    __shared__ _Float16 As[2][TBM * PA];        // [m][k]
    __shared__ _Float16 Bs[2][TBN * PA];        // [n][k] always

    const int z  = blockIdx.z;
    const int zb = z / Hdiv, zh = z % Hdiv;
    const _Float16* A16 = (const _Float16*)Aptr + (long long)zb * sAb + (long long)zh * sAh;
    const float*    A32 = (const float*)Aptr    + (long long)zb * sAb + (long long)zh * sAh;
    const _Float16* Bg  = Bh                    + (long long)zb * sBb + (long long)zh * sBh;
    float*    Cf = (float*)Cptr    + (long long)zb * sCb + (long long)zh * sCh;
    _Float16* Ch = (_Float16*)Cptr + (long long)zb * sCb + (long long)zh * sCh;

    const int m0 = blockIdx.x * TBM;
    const int n0 = blockIdx.y * TBN;
    const int t  = threadIdx.x;
    const int w  = t >> 5, l = t & 31;
    const int half = l >> 4, r = l & 15;
    const int wm = w % WROWS, wn = w / WROWS;

    v8f acc[MT][NT];
#pragma unroll
    for (int i = 0; i < MT; ++i)
#pragma unroll
        for (int j = 0; j < NT; ++j)
#pragma unroll
            for (int e = 0; e < 8; ++e) acc[i][j][e] = 0.0f;

    v8h ra16[CA16];
    v4f ra32[AF16 ? 1 : CA32];
    v8h rb[CB];

    // ---- global -> regs for K-step s ----
    auto loadreg = [&](int s) {
        int k0 = s << 5;
        if (AF16) {
#pragma unroll
            for (int c = 0; c < CA16; ++c) {
                int idx = t + c * 256;            // 0 .. TBM*4-1
                int m = idx >> 2, ko = (idx & 3) << 3;
                ra16[c] = *(const v8h*)&A16[(long long)(m0 + m) * lda + k0 + ko];
            }
        } else {
#pragma unroll
            for (int c = 0; c < CA32; ++c) {
                int idx = t + c * 256;            // 0 .. TBM*8-1
                int m = idx >> 3, ko = (idx & 7) << 2;
                ra32[c] = *(const v4f*)&A32[(long long)(m0 + m) * lda + k0 + ko];
            }
        }
#pragma unroll
        for (int c = 0; c < CB; ++c) {
            int idx = t + c * 256;                // 0 .. TBN*4-1
            if (BT) {
                int n = idx >> 2, ko = (idx & 3) << 3;
                rb[c] = *(const v8h*)&Bg[(long long)(n0 + n) * ldb + k0 + ko];
            } else {
                int k = idx / NB8, no = (idx % NB8) << 3;
                rb[c] = *(const v8h*)&Bg[(long long)(k0 + k) * ldb + n0 + no];
            }
        }
    };

    // ---- regs -> LDS buffer b ----
    auto storelds = [&](int b) {
        if (AF16) {
#pragma unroll
            for (int c = 0; c < CA16; ++c) {
                int idx = t + c * 256;
                int m = idx >> 2, ko = (idx & 3) << 3;
                *(v8h*)&As[b][m * PA + ko] = ra16[c];
            }
        } else {
#pragma unroll
            for (int c = 0; c < CA32; ++c) {
                int idx = t + c * 256;
                int m = idx >> 3, ko = (idx & 7) << 2;
                v4h h;
#pragma unroll
                for (int e = 0; e < 4; ++e) h[e] = (_Float16)ra32[c][e];
                *(v4h*)&As[b][m * PA + ko] = h;
            }
        }
#pragma unroll
        for (int c = 0; c < CB; ++c) {
            int idx = t + c * 256;
            if (BT) {
                int n = idx >> 2, ko = (idx & 3) << 3;
                *(v8h*)&Bs[b][n * PA + ko] = rb[c];
            } else {                               // transpose-scatter to [n][k]
                int k = idx / NB8, no = (idx % NB8) << 3;
#pragma unroll
                for (int e = 0; e < 8; ++e) Bs[b][(no + e) * PA + k] = rb[c][e];
            }
        }
    };

    // ---- fragment loads + 8 WMMAs from LDS buffer b ----
    auto compute = [&](int b) {
        Frag af[MT], bf[NT];
#pragma unroll
        for (int mt = 0; mt < MT; ++mt) {
            const _Float16* base = &As[b][((wm * MT + mt) * 16 + r) * PA];
#pragma unroll
            for (int i = 0; i < 8; ++i) {
                int kk = ((i >> 2) << 4) + (half << 3) + ((i & 3) << 1);
                af[mt].u[i] = *(const unsigned int*)&base[kk];
            }
        }
#pragma unroll
        for (int nt = 0; nt < NT; ++nt) {
            const _Float16* base = &Bs[b][((wn * NT + nt) * 16 + r) * PA];
#pragma unroll
            for (int i = 0; i < 8; ++i) {
                int kk = ((i >> 2) << 4) + (half << 3) + ((i & 3) << 1);
                bf[nt].u[i] = *(const unsigned int*)&base[kk];
            }
        }
#pragma unroll
        for (int mt = 0; mt < MT; ++mt)
#pragma unroll
            for (int nt = 0; nt < NT; ++nt)
                acc[mt][nt] = __builtin_amdgcn_wmma_f32_16x16x32_f16(
                    false, af[mt].h, false, bf[nt].h, (short)0, acc[mt][nt],
                    false, false);
    };

    const int nsteps = K >> 5;
    loadreg(0);
    storelds(0);
    __syncthreads();
    for (int s = 0; s < nsteps; ++s) {
        if (s + 1 < nsteps) loadreg(s + 1);
        compute(s & 1);
        if (s + 1 < nsteps) storelds((s + 1) & 1);
        __syncthreads();
    }

    // ---- epilogue: C/D 16x16 f32 layout (lane = N col, VGPR j -> M = 8*half+j) ----
#pragma unroll
    for (int mt = 0; mt < MT; ++mt)
#pragma unroll
        for (int nt = 0; nt < NT; ++nt) {
            int col = n0 + (wn * NT + nt) * 16 + r;
            float bs = bias ? bias[col] : 0.0f;
#pragma unroll
            for (int j = 0; j < 8; ++j) {
                int row = m0 + (wm * MT + mt) * 16 + (half << 3) + j;
                float v = acc[mt][nt][j] * alpha + bs;
                if (RELU) v = fmaxf(v, 0.0f);
                if (OUT16) Ch[(long long)row * ldc + col] = (_Float16)v;
                else       Cf[(long long)row * ldc + col] = v;
            }
        }
}

// fp32 -> f16 elementwise (vec4), for one-time weight conversion
__global__ __launch_bounds__(256) void cvt_kernel(const float* __restrict__ in,
                                                  _Float16* __restrict__ out, int n4)
{
    int i = blockIdx.x * 256 + threadIdx.x;
    if (i < n4) {
        v4f v = *(const v4f*)&in[i * 4];
        v4h h;
#pragma unroll
        for (int e = 0; e < 4; ++e) h[e] = (_Float16)v[e];
        *(v4h*)&out[i * 4] = h;
    }
}

// x = enc @ val_w + val_b + pos ; writes fp32 + f16 copies
__global__ __launch_bounds__(256) void embed_kernel(
    const float* __restrict__ enc, const float* __restrict__ vw,
    const float* __restrict__ vb, const float* __restrict__ pos,
    float* __restrict__ x, _Float16* __restrict__ xh)
{
    int idx = blockIdx.x * 256 + threadIdx.x;     // over 4096*512
    int d = idx & 511;
    int m = idx >> 9;
    int s = m & 511;
    const float* e = enc + m * 12;
    float acc = vb[d] + pos[s * 512 + d];
#pragma unroll
    for (int c = 0; c < 12; ++c) acc += e[c] * vw[c * 512 + d];
    x[idx] = acc;
    xh[idx] = (_Float16)acc;
}

// In-place row softmax over rows of length 512 (one block per row), fp32.
__global__ __launch_bounds__(256) void softmax_kernel(float* __restrict__ attn)
{
    long long row = blockIdx.x;
    float* p = attn + row * 512;
    int t = threadIdx.x;
    float a = p[t], b = p[t + 256];
    __shared__ float red[256];
    red[t] = fmaxf(a, b);
    __syncthreads();
    for (int st = 128; st > 0; st >>= 1) {
        if (t < st) red[t] = fmaxf(red[t], red[t + st]);
        __syncthreads();
    }
    float mx = red[0];
    __syncthreads();
    float ea = __expf(a - mx), eb = __expf(b - mx);
    red[t] = ea + eb;
    __syncthreads();
    for (int st = 128; st > 0; st >>= 1) {
        if (t < st) red[t] += red[t + st];
        __syncthreads();
    }
    float inv = 1.0f / red[0];
    p[t] = ea * inv;
    p[t + 256] = eb * inv;
}

// out = LayerNorm(y + xin)*g + b ; optional f16 copy (one block per D=512 row)
__global__ __launch_bounds__(256) void resid_ln_kernel(
    const float* __restrict__ y, const float* __restrict__ xin,
    const float* __restrict__ g, const float* __restrict__ b,
    float* __restrict__ out, _Float16* __restrict__ outh)
{
    long long row = blockIdx.x;
    int t = threadIdx.x;
    const float* yp = y + row * 512;
    const float* xp = xin + row * 512;
    float z0 = yp[t] + xp[t];
    float z1 = yp[t + 256] + xp[t + 256];
    __shared__ float s1[256], s2[256];
    s1[t] = z0 + z1;
    s2[t] = z0 * z0 + z1 * z1;
    __syncthreads();
    for (int st = 128; st > 0; st >>= 1) {
        if (t < st) { s1[t] += s1[t + st]; s2[t] += s2[t + st]; }
        __syncthreads();
    }
    float mean = s1[0] * (1.0f / 512.0f);
    float var  = s2[0] * (1.0f / 512.0f) - mean * mean;
    float rstd = rsqrtf(var + 1e-5f);
    float o0 = (z0 - mean) * rstd * g[t] + b[t];
    float o1 = (z1 - mean) * rstd * g[t + 256] + b[t + 256];
    out[row * 512 + t] = o0;
    out[row * 512 + t + 256] = o1;
    if (outh) {
        outh[row * 512 + t] = (_Float16)o0;
        outh[row * 512 + t + 256] = (_Float16)o1;
    }
}

extern "C" void kernel_launch(void* const* d_in, const int* in_sizes, int n_in,
                              void* d_out, int out_size, void* d_ws, size_t ws_size,
                              hipStream_t stream) {
    (void)in_sizes; (void)n_in; (void)out_size; (void)ws_size;
    const float* enc  = (const float*)d_in[0];
    const float* valw = (const float*)d_in[1];
    const float* valb = (const float*)d_in[2];
    const float* pos  = (const float*)d_in[3];
    const float* Wq   = (const float*)d_in[4];
    const float* bq   = (const float*)d_in[5];
    const float* Wk   = (const float*)d_in[6];
    const float* bk   = (const float*)d_in[7];
    const float* Wv   = (const float*)d_in[8];
    const float* bv   = (const float*)d_in[9];
    const float* Wo   = (const float*)d_in[10];
    const float* bo   = (const float*)d_in[11];
    const float* ln1g = (const float*)d_in[12];
    const float* ln1b = (const float*)d_in[13];
    const float* c1w  = (const float*)d_in[14];
    const float* c1b  = (const float*)d_in[15];
    const float* c2w  = (const float*)d_in[16];
    const float* c2b  = (const float*)d_in[17];
    const float* ln2g = (const float*)d_in[18];
    const float* ln2b = (const float*)d_in[19];

    constexpr int Bsz = 8, S = 512, D = 512, H = 8, F = 2048, L = 4;
    constexpr int MS = Bsz * S;                 // 4096 rows
    const float scale = 0.125f;                 // 1/sqrt(64)

    // -------- workspace layout --------
    float* xbuf = (float*)d_ws;                             // [4096,512] fp32
    float* ybuf = xbuf + (size_t)MS * D;                    // [4096,512] fp32
    _Float16* h16 = (_Float16*)(ybuf + (size_t)MS * D);
    _Float16* xh = h16;                  h16 += (size_t)MS * D;
    _Float16* qh = h16;                  h16 += (size_t)MS * D;
    _Float16* kh = h16;                  h16 += (size_t)MS * D;
    _Float16* vh = h16;                  h16 += (size_t)MS * D;
    _Float16* ch = h16;                  h16 += (size_t)MS * D;
    _Float16* hh = h16;                  h16 += (size_t)MS * F;
    _Float16* wq16 = h16;                h16 += (size_t)L * D * D;
    _Float16* wk16 = h16;                h16 += (size_t)L * D * D;
    _Float16* wv16 = h16;                h16 += (size_t)L * D * D;
    _Float16* wo16 = h16;                h16 += (size_t)L * D * D;
    _Float16* c1w16 = h16;               h16 += (size_t)L * D * F;
    _Float16* c2w16 = h16;               h16 += (size_t)L * F * D;

    float* outx = (float*)d_out;
    float* outa = outx + (size_t)MS * D;        // attention maps region

    // -------- one-time weight conversion to f16 --------
    {
        int n4 = L * D * D / 4;
        cvt_kernel<<<(n4 + 255) / 256, 256, 0, stream>>>(Wq, wq16, n4);
        cvt_kernel<<<(n4 + 255) / 256, 256, 0, stream>>>(Wk, wk16, n4);
        cvt_kernel<<<(n4 + 255) / 256, 256, 0, stream>>>(Wv, wv16, n4);
        cvt_kernel<<<(n4 + 255) / 256, 256, 0, stream>>>(Wo, wo16, n4);
        int n4f = L * D * F / 4;
        cvt_kernel<<<(n4f + 255) / 256, 256, 0, stream>>>(c1w, c1w16, n4f);
        cvt_kernel<<<(n4f + 255) / 256, 256, 0, stream>>>(c2w, c2w16, n4f);
    }

    embed_kernel<<<(MS * D) / 256, 256, 0, stream>>>(enc, valw, valb, pos, xbuf, xh);

    for (int l = 0; l < L; ++l) {
        const _Float16* wq_l = wq16 + (size_t)l * D * D;
        const _Float16* wk_l = wk16 + (size_t)l * D * D;
        const _Float16* wv_l = wv16 + (size_t)l * D * D;
        const _Float16* wo_l = wo16 + (size_t)l * D * D;
        const _Float16* w1_l = c1w16 + (size_t)l * D * F;
        const _Float16* w2_l = c2w16 + (size_t)l * F * D;
        const float* bq_l = bq + (size_t)l * D;
        const float* bk_l = bk + (size_t)l * D;
        const float* bv_l = bv + (size_t)l * D;
        const float* bo_l = bo + (size_t)l * D;
        const float* b1_l = c1b + (size_t)l * F;
        const float* b2_l = c2b + (size_t)l * D;
        const float* g1_l = ln1g + (size_t)l * D;  const float* gb1_l = ln1b + (size_t)l * D;
        const float* g2_l = ln2g + (size_t)l * D;  const float* gb2_l = ln2b + (size_t)l * D;
        float* attn_l = outa + (size_t)l * Bsz * H * S * S;

        // Q, K, V projections: f16 in -> f16 out (only consumed by GEMMs)
        dim3 gp(MS / 128, D / 128, 1);
        gemm2<128, 128, 4, false, false, true, true><<<gp, 256, 0, stream>>>(
            xh, wq_l, bq_l, qh, D, D, D, D, 0, 0, 0, 0, 0, 0, 1, 1.0f);
        gemm2<128, 128, 4, false, false, true, true><<<gp, 256, 0, stream>>>(
            xh, wk_l, bk_l, kh, D, D, D, D, 0, 0, 0, 0, 0, 0, 1, 1.0f);
        gemm2<128, 128, 4, false, false, true, true><<<gp, 256, 0, stream>>>(
            xh, wv_l, bv_l, vh, D, D, D, D, 0, 0, 0, 0, 0, 0, 1, 1.0f);

        // scores = scale * Q @ K^T per (b,h), fp32 straight into d_out
        {
            dim3 gs(S / 128, S / 128, Bsz * H);
            gemm2<128, 128, 4, true, false, true, false><<<gs, 256, 0, stream>>>(
                qh, kh, nullptr, attn_l,
                64, D, D, S,
                (long long)S * D, 64LL,
                (long long)S * D, 64LL,
                (long long)H * S * S, (long long)S * S,
                H, scale);
        }

        softmax_kernel<<<Bsz * H * S, 256, 0, stream>>>(attn_l);

        // ctx = attn @ V per (b,h): A fp32 from d_out, B f16, out f16
        {
            dim3 gc(S / 256, 1, Bsz * H);
            gemm2<256, 64, 8, false, false, false, true><<<gc, 256, 0, stream>>>(
                attn_l, vh, nullptr, ch,
                S, S, D, D,
                (long long)H * S * S, (long long)S * S,
                (long long)S * D, 64LL,
                (long long)S * D, 64LL,
                H, 1.0f);
        }

        // output projection (fp32 out for residual) + residual LN1
        gemm2<128, 128, 4, false, false, true, false><<<gp, 256, 0, stream>>>(
            ch, wo_l, bo_l, ybuf, D, D, D, D, 0, 0, 0, 0, 0, 0, 1, 1.0f);
        resid_ln_kernel<<<MS, 256, 0, stream>>>(ybuf, xbuf, g1_l, gb1_l, xbuf, xh);

        // FFN: relu(x @ c1_w + c1_b) -> f16 hidden; @ c2_w + c2_b -> fp32; LN2
        dim3 gf1(MS / 128, F / 128, 1);
        gemm2<128, 128, 4, false, true, true, true><<<gf1, 256, 0, stream>>>(
            xh, w1_l, b1_l, hh, D, D, F, F, 0, 0, 0, 0, 0, 0, 1, 1.0f);
        gemm2<128, 128, 4, false, false, true, false><<<gp, 256, 0, stream>>>(
            hh, w2_l, b2_l, ybuf, F, F, D, D, 0, 0, 0, 0, 0, 0, 1, 1.0f);

        float* lnout = (l == L - 1) ? outx : xbuf;
        _Float16* lnouth = (l == L - 1) ? nullptr : xh;
        resid_ln_kernel<<<MS, 256, 0, stream>>>(ybuf, xbuf, g2_l, gb2_l, lnout, lnouth);
    }
}
